// Router_58222576665019
// MI455X (gfx1250) — matrix operand order repaired and checked by reading, not compiled, not run on hardware
//
#include <hip/hip_runtime.h>
#include <hip/hip_bf16.h>

#define NUM_EXPERTS 64
#define TOP_K 8
#define KC 128                  // K-chunk per LDS stage
#define MT 64                   // tokens per block
#define NTHREADS 512            // 16 wave32: 4 M-tiles x 4 N-tiles
#define ASTR (KC + 8)           // padded LDS row stride (bf16 elems) -> 272B, 16B aligned
#define BSTR (KC + 8)
#define LSTR (NUM_EXPERTS + 4)  // padded logits row stride (f32)

typedef __attribute__((ext_vector_type(16))) __bf16 v16bf;
typedef __attribute__((ext_vector_type(8)))  float  v8f;

union Frag {
    v16bf v;
    uint4 q[2];
};

__device__ __forceinline__ unsigned pack_bf16(float a, float b) {
    unsigned short lo = __builtin_bit_cast(unsigned short, (__bf16)a);
    unsigned short hi = __builtin_bit_cast(unsigned short, (__bf16)b);
    return (unsigned)lo | ((unsigned)hi << 16);
}

__global__ void zero_counts_kernel(int* __restrict__ counts) {
    counts[threadIdx.x] = 0;
}

// Block: 512 threads (16 wave32). Block owns 64 tokens.
// Wave w: M-tile (w>>2)*16 tokens, N-tile (w&3)*16 experts.
__global__ void __launch_bounds__(NTHREADS)
router_topk_kernel(const float* __restrict__ x,
                   const float* __restrict__ gate_w,
                   const float* __restrict__ expert_bias,
                   float* __restrict__ scores,
                   int*   __restrict__ indices,
                   int*   __restrict__ counts,
                   int D) {
    __shared__ __align__(16) __bf16 ldsA[2][MT * ASTR];           // token tiles (bf16)
    __shared__ __align__(16) __bf16 ldsB[2][NUM_EXPERTS * BSTR];  // gate tiles  (bf16)
    __shared__ float ldsL[MT * LSTR];                             // logits tile (f32)
    __shared__ float ldsBias[NUM_EXPERTS];
    __shared__ int   ldsCnt[NUM_EXPERTS];

    const int  tid   = threadIdx.x;
    const int  lane  = tid & 31;
    const int  wave  = tid >> 5;
    const int  waveM = wave >> 2;               // 0..3
    const int  waveN = wave & 3;                // 0..3
    const long tokBase = (long)blockIdx.x * MT;

    if (tid < NUM_EXPERTS) { ldsBias[tid] = expert_bias[tid]; ldsCnt[tid] = 0; }

    v8f acc = {0.f, 0.f, 0.f, 0.f, 0.f, 0.f, 0.f, 0.f};

    const int r  = lane & 15;                   // A: row-in-tile / B: col-in-tile
    const int kb = (lane & 16) ? 8 : 0;         // per-lane-half K base (ISA 16-bit layout)
    const int aRow = waveM * 16 + r;            // token row within block tile
    const int bRow = waveN * 16 + r;            // expert row

    // ---------------- prologue: stage chunk 0 into buffer 0 ----------------
    {
        #pragma unroll
        for (int j = 0; j < 4; ++j) {           // MT*KC/4 f32 = 2048 float4 = 4*512
            int i   = j * NTHREADS + tid;
            int row = i >> 5;                   // KC/4 == 32 float4 per row
            int c4  = i & 31;
            float4 a = *(const float4*)(x + (tokBase + row) * D + 4 * c4);
            float4 b = *(const float4*)(gate_w + (long)row * D + 4 * c4);
            *(uint2*)&ldsA[0][row * ASTR + 4 * c4] =
                make_uint2(pack_bf16(a.x, a.y), pack_bf16(a.z, a.w));
            *(uint2*)&ldsB[0][row * BSTR + 4 * c4] =
                make_uint2(pack_bf16(b.x, b.y), pack_bf16(b.z, b.w));
        }
    }
    __syncthreads();

    // ---------------- main K loop: double-buffered ----------------
    int buf = 0;
    for (int kc0 = 0; kc0 < D; kc0 += KC) {
        const bool has_next = (kc0 + KC) < D;
        float4 ra[4], rb[4];
        if (has_next) {                          // issue next chunk's loads early
            #pragma unroll
            for (int j = 0; j < 4; ++j) {
                int i   = j * NTHREADS + tid;
                int row = i >> 5;
                int c4  = i & 31;
                ra[j] = *(const float4*)(x + (tokBase + row) * D + (kc0 + KC) + 4 * c4);
                rb[j] = *(const float4*)(gate_w + (long)row * D + (kc0 + KC) + 4 * c4);
            }
        }

        // ---- 4 WMMAs on current buffer: D = A(16x32 bf16) x B(32x16 bf16) + D ----
        const __bf16* bA = &ldsA[buf][aRow * ASTR + kb];
        const __bf16* bB = &ldsB[buf][bRow * BSTR + kb];
        #pragma unroll
        for (int kk = 0; kk < KC; kk += 32) {
            Frag a, b;
            a.q[0] = *(const uint4*)(bA + kk);        // K = kb .. kb+7
            a.q[1] = *(const uint4*)(bA + kk + 16);   // K = kb+16 .. kb+23
            b.q[0] = *(const uint4*)(bB + kk);
            b.q[1] = *(const uint4*)(bB + kk + 16);
            acc = __builtin_amdgcn_wmma_f32_16x16x32_bf16(
                false, a.v, false, b.v, (short)0, acc, false, false);
        }

        if (has_next) {                          // convert + store into other buffer
            #pragma unroll
            for (int j = 0; j < 4; ++j) {
                int i   = j * NTHREADS + tid;
                int row = i >> 5;
                int c4  = i & 31;
                *(uint2*)&ldsA[buf ^ 1][row * ASTR + 4 * c4] =
                    make_uint2(pack_bf16(ra[j].x, ra[j].y), pack_bf16(ra[j].z, ra[j].w));
                *(uint2*)&ldsB[buf ^ 1][row * BSTR + 4 * c4] =
                    make_uint2(pack_bf16(rb[j].x, rb[j].y), pack_bf16(rb[j].z, rb[j].w));
            }
        }
        buf ^= 1;
        __syncthreads();
    }

    // ---- write logits (+ routing bias): C layout VGPR rr -> M = rr + 8*(lane>=16) ----
    {
        const int e   = waveN * 16 + r;
        const int mhi = (lane & 16) ? 8 : 0;
        const float be = ldsBias[e];
        #pragma unroll
        for (int rr = 0; rr < 8; ++rr)
            ldsL[(waveM * 16 + rr + mhi) * LSTR + e] = acc[rr] + be;
    }
    __syncthreads();

    // ---- per-token top-8, softmax, outputs, LDS histogram (1 lane per token) ----
    if (tid < MT) {
        const float* row = &ldsL[tid * LSTR];
        unsigned long long used = 0ull;
        float vals[TOP_K];
        int   idx[TOP_K];
        for (int j = 0; j < TOP_K; ++j) {
            float best = -3.402823466e38f;
            int   bi   = 0;
            for (int e = 0; e < NUM_EXPERTS; ++e) {
                if ((used >> e) & 1ull) continue;
                float v = row[e];
                if (v > best) { best = v; bi = e; }   // strict > == lowest-index tiebreak
            }
            used |= (1ull << bi);
            vals[j] = best;
            idx[j]  = bi;
        }
        float m = vals[0];                      // sorted descending
        float p[TOP_K], sum = 0.f;
        #pragma unroll
        for (int j = 0; j < TOP_K; ++j) { p[j] = __expf(vals[j] - m); sum += p[j]; }
        float inv = 1.0f / sum;
        long t = tokBase + tid;
        #pragma unroll
        for (int j = 0; j < TOP_K; ++j) {
            scores[t * TOP_K + j]  = p[j] * inv;
            indices[t * TOP_K + j] = idx[j];
            atomicAdd(&ldsCnt[idx[j]], 1);
        }
    }
    __syncthreads();
    if (tid < NUM_EXPERTS) atomicAdd(&counts[tid], ldsCnt[tid]);
}

extern "C" void kernel_launch(void* const* d_in, const int* in_sizes, int n_in,
                              void* d_out, int out_size, void* d_ws, size_t ws_size,
                              hipStream_t stream) {
    const float* x    = (const float*)d_in[0];
    const float* gw   = (const float*)d_in[1];
    const float* bias = (const float*)d_in[2];

    const int D      = in_sizes[1] / NUM_EXPERTS;   // 2048
    const int tokens = in_sizes[0] / D;             // 16384

    // d_out layout (flat, return order): scores f32 | indices i32 | counts i32
    float* scores  = (float*)d_out;
    int*   indices = (int*)d_out + (size_t)tokens * TOP_K;
    int*   counts  = (int*)d_out + (size_t)tokens * TOP_K * 2;

    zero_counts_kernel<<<1, NUM_EXPERTS, 0, stream>>>(counts);

    const int blocks = tokens / MT;
    router_topk_kernel<<<blocks, NTHREADS, 0, stream>>>(x, gw, bias, scores, indices,
                                                        counts, D);
}
